// MultiHeadSelfAttention_32427003085154
// MI455X (gfx1250) — compile-verified, matrix operand order checked by reading, not run on hardware
//
#include <hip/hip_runtime.h>
#include <stdint.h>

#define B_  2
#define S_  2048
#define D_  1024
#define H_  16
#define DH_ 64

typedef __bf16 bf16_t;
typedef __attribute__((ext_vector_type(16))) __bf16 v16bf;
typedef __attribute__((ext_vector_type(8)))  __bf16 v8bf;
typedef __attribute__((ext_vector_type(2)))  __bf16 v2bf;
typedef __attribute__((ext_vector_type(8)))  float  v8f;
typedef __attribute__((ext_vector_type(4)))  int    v4i;

// ---------- feature probes (compile-safe) ----------
#if __has_builtin(__builtin_amdgcn_global_load_async_to_lds_b128)
#define HAVE_ASYNC_LDS 1
#else
#define HAVE_ASYNC_LDS 0
#endif

#if HAVE_ASYNC_LDS
typedef __attribute__((address_space(1))) v4i GV4;   // global int4
typedef __attribute__((address_space(3))) v4i LV4;   // LDS int4
#endif

// fp32 -> bf16, round-half-up (1-ulp-of-bf16 bias; cheap: 1 add + hi16)
__device__ __forceinline__ bf16_t f2bf(float f) {
  union { float f; uint32_t u; } a; a.f = f;
  union { uint16_t s; bf16_t b; } o;
  o.s = (uint16_t)((a.u + 0x8000u) >> 16);
  return o.b;
}
// two fp32 -> packed bf16x2 in one dword
__device__ __forceinline__ uint32_t pack2(float x, float y) {
#if __has_builtin(__builtin_amdgcn_cvt_pk_bf16_f32)
  union { v2bf v; uint32_t u; } o;
  o.v = __builtin_amdgcn_cvt_pk_bf16_f32(x, y);
  return o.u;
#else
  union { float f; uint32_t u; } ax, ay; ax.f = x; ay.f = y;
  uint32_t xu = ax.u + 0x8000u;
  uint32_t yu = ay.u + 0x8000u;
#if __has_builtin(__builtin_amdgcn_perm)
  return __builtin_amdgcn_perm(yu, xu, 0x07060302u);  // {y[31:16], x[31:16]}
#else
  return (xu >> 16) | (yu & 0xFFFF0000u);
#endif
#endif
}

// 16-byte global -> LDS copy; async (ASYNCcnt) when the toolchain exposes it.
__device__ __forceinline__ void copy16(const bf16_t* g, bf16_t* l) {
#if HAVE_ASYNC_LDS
  __builtin_amdgcn_global_load_async_to_lds_b128((GV4*)(uintptr_t)g, (LV4*)l, 0, 0);
#else
  *(uint4*)l = *(const uint4*)g;
#endif
}
__device__ __forceinline__ void wait_async() {
#if HAVE_ASYNC_LDS
#if __has_builtin(__builtin_amdgcn_s_wait_asynccnt)
  __builtin_amdgcn_s_wait_asynccnt(0);
#else
  asm volatile("s_wait_asynccnt 0" ::: "memory");
#endif
#endif
}

constexpr int BM  = 128;
constexpr int BN  = 64;
constexpr int BK  = 64;
constexpr int LDT = BK + 8;   // padded LDS row stride (bf16); 144B rows keep 16B align

// Fragment load matching the CDNA5 16-bit A/B operand layout:
// lanes 0-15: row = base+lane, K {0..7, 16..23}; lanes 16-31: row = base+lane-16, K {8..15, 24..31}
__device__ __forceinline__ v16bf load_frag(const bf16_t* s, int row_base, int kk, int lane) {
  int r  = row_base + (lane & 15);
  int ko = (lane & 16) ? 8 : 0;
  const bf16_t* p = s + r * LDT + kk + ko;
  v8bf lo = *(const v8bf*)(p);        // ds_load_b128
  v8bf hi = *(const v8bf*)(p + 16);   // ds_load_b128
  v16bf o;
#pragma unroll
  for (int i = 0; i < 8; ++i) { o[i] = lo[i]; o[i + 8] = hi[i]; }
  return o;
}

// C[M,N] = alpha * A[M,K] * B[N,K]^T  (+bias[N]) (+mask->-inf), fp32 accumulate via WMMA bf16.
// Batched over blockIdx.z with per-(batch,head) offsets: off = (z/heads)*s?b + (z%heads)*s?h.
// BTRANS: B source is stored K-major (element (n,k) at src[k*ldb + n]) -> NN GEMM.
template<bool AF32, bool BF32, bool BTRANS, bool OUT_BF16, bool BIAS, bool MASK>
__global__ __launch_bounds__(256) void gemm_nt(
    const void* __restrict__ Ap, const void* __restrict__ Bp,
    const float* __restrict__ bias, const int* __restrict__ maskp,
    void* __restrict__ Cp,
    int M, int N, int K, int lda, int ldb, int ldc, int ldm,
    long sAb, long sAh, long sBb, long sBh, long sCb, long sCh, long sMb,
    int heads, float alpha)
{
  __shared__ bf16_t As[BM * LDT];
  __shared__ bf16_t Bs[BN * LDT];

  const int z  = blockIdx.z;
  const int bb = z / heads, hh = z % heads;
  const long aoff = sAb * bb + sAh * hh;
  const long boff = sBb * bb + sBh * hh;
  const long coff = sCb * bb + sCh * hh;

  const int m0   = blockIdx.y * BM;
  const int n0   = blockIdx.x * BN;
  const int t    = threadIdx.x;
  const int lane = t & 31;
  const int w    = t >> 5;
  const int wm   = (w & 3) * 32;   // wave M offset: 0,32,64,96
  const int wn   = (w >> 2) * 32;  // wave N offset: 0,32

  v8f acc[2][2] = {};

  for (int kt = 0; kt < K; kt += BK) {
    __syncthreads();
    // ---- stage A tile (BM x BK) into LDS as bf16 ----
    if (AF32) {
      const float* A = (const float*)Ap + aoff;
      float4 ta[8];
#pragma unroll
      for (int i = 0; i < 8; ++i) {            // batch the 8 global_load_b128 first
        int idx = t + i * 256;
        int r = idx >> 4;
        int c = (idx & 15) << 2;
        ta[i] = *(const float4*)(A + (long)(m0 + r) * lda + kt + c);
      }
#pragma unroll
      for (int i = 0; i < 8; ++i) {            // then convert + ds_store
        int idx = t + i * 256;
        int r = idx >> 4;
        int c = (idx & 15) << 2;
        uint2 pk; pk.x = pack2(ta[i].x, ta[i].y); pk.y = pack2(ta[i].z, ta[i].w);
        *(uint2*)(&As[r * LDT + c]) = pk;
      }
      if (kt + BK < K)
        __builtin_prefetch((const void*)(A + (long)(m0 + (t >> 1)) * lda + kt + BK), 0, 1);
    } else {
      const bf16_t* A = (const bf16_t*)Ap + aoff;
#pragma unroll
      for (int i = 0; i < 4; ++i) {            // async global -> LDS movers
        int idx = t + i * 256;
        int r = idx >> 3;
        int c = (idx & 7) << 3;
        copy16(A + (long)(m0 + r) * lda + kt + c, &As[r * LDT + c]);
      }
      if (kt + BK < K)
        __builtin_prefetch((const void*)(A + (long)(m0 + (t >> 1)) * lda + kt + BK), 0, 1);
    }
    // ---- stage B tile (BN x BK) into LDS as bf16, row n / col k ----
    if (BF32) {
      const float* Bm = (const float*)Bp + boff;
      float4 tb[4];
#pragma unroll
      for (int i = 0; i < 4; ++i) {
        int idx = t + i * 256;
        int r = idx >> 4;
        int c = (idx & 15) << 2;
        tb[i] = *(const float4*)(Bm + (long)(n0 + r) * ldb + kt + c);
      }
#pragma unroll
      for (int i = 0; i < 4; ++i) {
        int idx = t + i * 256;
        int r = idx >> 4;
        int c = (idx & 15) << 2;
        uint2 pk; pk.x = pack2(tb[i].x, tb[i].y); pk.y = pack2(tb[i].z, tb[i].w);
        *(uint2*)(&Bs[r * LDT + c]) = pk;
      }
    } else if (BTRANS) {
      // source is K-major: element (n,k) at src[k*ldb + n]; read 8 contiguous n, scatter
      const bf16_t* Bm = (const bf16_t*)Bp + boff;
      uint4 tb[2];
#pragma unroll
      for (int i = 0; i < 2; ++i) {
        int idx = t + i * 256;
        int kk = idx >> 3;
        int nn = (idx & 7) << 3;
        tb[i] = *(const uint4*)(Bm + (long)(kt + kk) * ldb + n0 + nn);
      }
#pragma unroll
      for (int i = 0; i < 2; ++i) {
        int idx = t + i * 256;
        int kk = idx >> 3;                // 0..63
        int nn = (idx & 7) << 3;          // 0..56
        const bf16_t* e = (const bf16_t*)&tb[i];
#pragma unroll
        for (int u = 0; u < 8; ++u) Bs[(nn + u) * LDT + kk] = e[u];
      }
    } else {
      const bf16_t* Bm = (const bf16_t*)Bp + boff;
#pragma unroll
      for (int i = 0; i < 2; ++i) {            // async global -> LDS movers
        int idx = t + i * 256;
        int r = idx >> 3;
        int c = (idx & 7) << 3;
        copy16(Bm + (long)(n0 + r) * ldb + kt + c, &Bs[r * LDT + c]);
      }
    }
    // async tile movers (if used) must land before the barrier
    if ((!AF32) || (!BF32 && !BTRANS)) wait_async();
    __syncthreads();

#pragma unroll
    for (int kk = 0; kk < BK; kk += 32) {
      v16bf a0 = load_frag(As, wm,      kk, lane);
      v16bf a1 = load_frag(As, wm + 16, kk, lane);
      v16bf b0 = load_frag(Bs, wn,      kk, lane);
      v16bf b1 = load_frag(Bs, wn + 16, kk, lane);
      acc[0][0] = __builtin_amdgcn_wmma_f32_16x16x32_bf16(false, a0, false, b0, (short)0, acc[0][0], false, false);
      acc[0][1] = __builtin_amdgcn_wmma_f32_16x16x32_bf16(false, a0, false, b1, (short)0, acc[0][1], false, false);
      acc[1][0] = __builtin_amdgcn_wmma_f32_16x16x32_bf16(false, a1, false, b0, (short)0, acc[1][0], false, false);
      acc[1][1] = __builtin_amdgcn_wmma_f32_16x16x32_bf16(false, a1, false, b1, (short)0, acc[1][1], false, false);
    }
  }

  // ---- epilogue: C/D layout -> VGPR j holds M=j (lanes 0-15) / M=j+8 (lanes 16-31), N=lane&15
  const int rb = m0 + wm + ((lane & 16) ? 8 : 0);
  const int cb = n0 + wn + (lane & 15);
#pragma unroll
  for (int i = 0; i < 2; ++i) {
#pragma unroll
    for (int j = 0; j < 2; ++j) {
#pragma unroll
      for (int e = 0; e < 8; ++e) {
        long row = rb + i * 16 + e;
        long col = cb + j * 16;
        float val = acc[i][j][e] * alpha;
        if (BIAS) val += bias[col];
        if (MASK) {
          if (maskp[sMb * bb + row * (long)ldm + col] == 0) val = -__builtin_inff();
        }
        if (OUT_BF16) ((bf16_t*)Cp)[coff + row * (long)ldc + col] = f2bf(val);
        else          ((float*)Cp)[coff + row * (long)ldc + col] = val;
      }
    }
  }
}

// One 256-thread block per row of 2048 fp32 scores; in-place masked softmax.
__global__ __launch_bounds__(256) void softmax_rows(float* __restrict__ Wt) {
  const long row = blockIdx.x;
  float* p = Wt + row * (long)S_;
  const int t = threadIdx.x;
  __shared__ float red[8];
  __shared__ float red2[8];

  float v[8];
  float m = -__builtin_inff();
#pragma unroll
  for (int i = 0; i < 8; ++i) { v[i] = p[t + i * 256]; m = fmaxf(m, v[i]); }
#pragma unroll
  for (int off = 16; off > 0; off >>= 1) m = fmaxf(m, __shfl_xor(m, off, 32));
  if ((t & 31) == 0) red[t >> 5] = m;
  __syncthreads();
#pragma unroll
  for (int kk = 0; kk < 8; ++kk) m = fmaxf(m, red[kk]);

  float s = 0.f;
#pragma unroll
  for (int i = 0; i < 8; ++i) { v[i] = __expf(v[i] - m); s += v[i]; }
#pragma unroll
  for (int off = 16; off > 0; off >>= 1) s += __shfl_xor(s, off, 32);
  if ((t & 31) == 0) red2[t >> 5] = s;
  __syncthreads();
  float tot = 0.f;
#pragma unroll
  for (int kk = 0; kk < 8; ++kk) tot += red2[kk];
  float inv = 1.0f / tot;
#pragma unroll
  for (int i = 0; i < 8; ++i) p[t + i * 256] = v[i] * inv;
}

extern "C" void kernel_launch(void* const* d_in, const int* in_sizes, int n_in,
                              void* d_out, int out_size, void* d_ws, size_t ws_size,
                              hipStream_t stream) {
  const float* q    = (const float*)d_in[0];
  const float* k    = (const float*)d_in[1];
  const float* v    = (const float*)d_in[2];
  const int*   mask = (const int*)  d_in[3];
  const float* wq_w = (const float*)d_in[4];
  const float* wq_b = (const float*)d_in[5];
  const float* wk_w = (const float*)d_in[6];
  const float* wk_b = (const float*)d_in[7];
  const float* wv_w = (const float*)d_in[8];
  const float* wv_b = (const float*)d_in[9];
  const float* dw   = (const float*)d_in[10];
  const float* db   = (const float*)d_in[11];

  float* outp = (float*)d_out;                       // [B,S,D] fp32
  float* wout = outp + (long)B_ * S_ * D_;           // [B,H,S,S] fp32 softmax weights

  char* ws = (char*)d_ws;
  const size_t QKV_BYTES = (size_t)B_ * S_ * D_ * sizeof(bf16_t);  // 8 MB each
  bf16_t* Qb = (bf16_t*)(ws);
  bf16_t* Kb = (bf16_t*)(ws + QKV_BYTES);
  bf16_t* Vb = (bf16_t*)(ws + 2 * QKV_BYTES);
  bf16_t* Ao = (bf16_t*)(ws + 3 * QKV_BYTES);        // merged-head attn output, bf16

  const long SD  = (long)S_ * D_;       // 2097152
  const long SSl = (long)S_ * S_;       // 4194304
  const long HSS = (long)H_ * SSl;      // 67108864
  const float scale = 0.125f;           // 1/sqrt(64)
  dim3 blk(256);

  // --- QKV projections: [4096,1024] @ W^T + b -> bf16 ---
  gemm_nt<true, true, false, true, true, false><<<dim3(D_ / BN, (B_ * S_) / BM, 1), blk, 0, stream>>>(
      q, wq_w, wq_b, nullptr, Qb, B_ * S_, D_, D_, D_, D_, D_, 0,
      0, 0, 0, 0, 0, 0, 0, 1, 1.0f);
  gemm_nt<true, true, false, true, true, false><<<dim3(D_ / BN, (B_ * S_) / BM, 1), blk, 0, stream>>>(
      k, wk_w, wk_b, nullptr, Kb, B_ * S_, D_, D_, D_, D_, D_, 0,
      0, 0, 0, 0, 0, 0, 0, 1, 1.0f);
  gemm_nt<true, true, false, true, true, false><<<dim3(D_ / BN, (B_ * S_) / BM, 1), blk, 0, stream>>>(
      v, wv_w, wv_b, nullptr, Vb, B_ * S_, D_, D_, D_, D_, D_, 0,
      0, 0, 0, 0, 0, 0, 0, 1, 1.0f);

  // --- scores = scale * Qh @ Kh^T, masked -> fp32 weights region of d_out ---
  gemm_nt<false, false, false, false, false, true><<<dim3(S_ / BN, S_ / BM, B_ * H_), blk, 0, stream>>>(
      Qb, Kb, nullptr, mask, wout, S_, S_, DH_, D_, D_, S_, S_,
      SD, DH_, SD, DH_, HSS, SSl, SSl, H_, scale);

  // --- row softmax in place (B*H*S rows of 2048) ---
  softmax_rows<<<dim3(B_ * H_ * S_), blk, 0, stream>>>(wout);

  // --- attn out = P @ Vh  (NN via BTRANS; A fp32 probs converted on LDS stage) ---
  gemm_nt<true, false, true, true, false, false><<<dim3(DH_ / BN, S_ / BM, B_ * H_), blk, 0, stream>>>(
      wout, Vb, nullptr, nullptr, Ao, S_, DH_, S_, S_, D_, D_, 0,
      HSS, SSl, SD, DH_, SD, DH_, 0, H_, 1.0f);

  // --- dense: [4096,1024] @ dense_w^T + b -> fp32 d_out ---
  gemm_nt<false, true, false, false, true, false><<<dim3(D_ / BN, (B_ * S_) / BM, 1), blk, 0, stream>>>(
      Ao, dw, db, nullptr, outp, B_ * S_, D_, D_, D_, D_, D_, 0,
      0, 0, 0, 0, 0, 0, 0, 1, 1.0f);
}
